// PosUpdate_48507360641323
// MI455X (gfx1250) — matrix operand order repaired and checked by reading, not compiled, not run on hardware
//
#include <hip/hip_runtime.h>
#include <math.h>

typedef __attribute__((ext_vector_type(16))) __bf16 v16bf;
typedef __attribute__((ext_vector_type(8)))  float  v8f;
typedef __attribute__((ext_vector_type(4)))  float  v4f;

union BFrag { v16bf v; __bf16 b[16]; };

// fast SiLU: h * rcp(1 + exp(-h)); exp/rcp are TRANS-pipe ops that co-execute
// with VALU and XDL WMMA.
__device__ __forceinline__ float fast_silu(float h) {
    float e = __expf(-h);
    return h * __builtin_amdgcn_rcpf(1.0f + e);
}

// ---------------------------------------------------------------------------
// Kernel 1: repack W1 of both MLPs (row-major [K=128][N=128] f32) into the
// wave32 WMMA B-operand layout as bf16.
// Fragment (mlp, kc, nt): B tile is 32x16; lane l supplies N = nt*16 + (l%16),
// K = kc*32 + (l/16)*8 + j (j<8), kc*32 + 16 + (l/16)*8 + (j-8) (j>=8) (2 bf16 per VGPR, 8 VGPRs).
// Buffer layout: bufB[((mlp*4+kc)*8+nt)*32 + lane][16]  (32B contiguous/lane).
// ---------------------------------------------------------------------------
__global__ void pack_weights_kernel(const float* __restrict__ Ww1,
                                    const float* __restrict__ Wa1,
                                    __bf16* __restrict__ bufB) {
    int t = blockIdx.x * blockDim.x + threadIdx.x;   // 2048 threads total
    if (t >= 2048) return;
    int lane = t & 31;
    int nt   = (t >> 5) & 7;
    int kc   = (t >> 8) & 3;
    int m    = t >> 10;
    const float* W = m ? Wa1 : Ww1;
    int lm = lane & 15, half = lane >> 4;
    int n = nt * 16 + lm;
    __bf16* o = bufB + (size_t)t * 16;
#pragma unroll
    for (int j = 0; j < 16; ++j) {
        int k = kc * 32 + (j < 8 ? half * 8 + j : 16 + half * 8 + (j - 8));
        o[j] = (__bf16)W[(size_t)k * 128 + n];
    }
}

// ---------------------------------------------------------------------------
// Kernel 2: out = x ; a_sum = 0
// ---------------------------------------------------------------------------
__global__ void init_kernel(const float* __restrict__ x, float* __restrict__ out,
                            float* __restrict__ a_sum, int n3, int n) {
    int i = blockIdx.x * blockDim.x + threadIdx.x;
    if (i < n3) out[i] = x[i];
    if (i < n)  a_sum[i] = 0.0f;
}

// ---------------------------------------------------------------------------
// Kernel 3: fused edge MLPs via WMMA. One wave handles 16 edges.
// hidden = SiLU(m_ij @ W1 + b1); scalar = hidden @ W2 + b2 for both MLPs.
// MLP0 -> w_raw[e]; MLP1 -> a_ij[e] = exp(.), atomic a_sum[src] += a_ij.
// ---------------------------------------------------------------------------
__global__ void __launch_bounds__(256)
edge_mlp_kernel(const float* __restrict__ m_ij,
                const __bf16* __restrict__ bufB,
                const float* __restrict__ bw1, const float* __restrict__ Ww2,
                const float* __restrict__ bw2,
                const float* __restrict__ ba1, const float* __restrict__ Wa2,
                const float* __restrict__ ba2,
                const long long* __restrict__ edges,
                float* __restrict__ w_raw, float* __restrict__ a_ij,
                float* __restrict__ a_sum, int nTiles) {
    const int lane = threadIdx.x & 31;
    const int gwid = blockIdx.x * (blockDim.x >> 5) + (threadIdx.x >> 5);
    if (gwid >= nTiles) return;                    // wave-uniform: EXEC stays all-ones
    const int tileBase = gwid << 4;
    const int lm = lane & 15, half = lane >> 4;

    // --- load A fragments: 16 edges x 128 K, bf16, documented 16-bit A 16x32 layout.
    // lane l (M = l%16): j=0..7  -> K = kc*32 + half*8 + j
    //                    j=8..15 -> K = kc*32 + 16 + half*8 + (j-8)
    const v4f* rp4 = (const v4f*)(m_ij + (size_t)(tileBase + lm) * 128);
    BFrag afrag[4];
#pragma unroll
    for (int kc = 0; kc < 4; ++kc) {
        const int o = kc * 8 + half * 2;           // float4 units
        v4f p0 = __builtin_nontemporal_load(&rp4[o]);
        v4f p1 = __builtin_nontemporal_load(&rp4[o + 1]);
        v4f p2 = __builtin_nontemporal_load(&rp4[o + 4]);
        v4f p3 = __builtin_nontemporal_load(&rp4[o + 5]);
        float f[16] = { p0.x, p0.y, p0.z, p0.w, p1.x, p1.y, p1.z, p1.w,
                        p2.x, p2.y, p2.z, p2.w, p3.x, p3.y, p3.z, p3.w };
#pragma unroll
        for (int j = 0; j < 16; ++j) afrag[kc].b[j] = (__bf16)f[j];  // native cvt (pk) path
    }

    const v16bf* bp = (const v16bf*)bufB;

#pragma unroll
    for (int mlp = 0; mlp < 2; ++mlp) {
        const float* b1 = mlp ? ba1 : bw1;
        const float* W2 = mlp ? Wa2 : Ww2;
        float partial[8] = {0.f, 0.f, 0.f, 0.f, 0.f, 0.f, 0.f, 0.f};

#pragma unroll
        for (int nt = 0; nt < 8; ++nt) {
            v8f acc = {0.f, 0.f, 0.f, 0.f, 0.f, 0.f, 0.f, 0.f};
#pragma unroll
            for (int kc = 0; kc < 4; ++kc) {
                v16bf b = bp[(size_t)(((mlp * 4 + kc) * 8 + nt) * 32 + lane)];
                acc = __builtin_amdgcn_wmma_f32_16x16x32_bf16(
                          false, afrag[kc].v, false, b, (short)0, acc, false, false);
            }
            // C/D layout: lane l, VGPR r -> edge M = r + 8*half, hidden N = nt*16 + lm
            const int n = nt * 16 + lm;
            const float bias = b1[n];
            const float w2   = W2[n];
#pragma unroll
            for (int r = 0; r < 8; ++r) {
                partial[r] += fast_silu(acc[r] + bias) * w2;
            }
        }

        // reduce the 16-lane hidden dimension (offsets <16 stay within each half)
#pragma unroll
        for (int off = 1; off < 16; off <<= 1)
#pragma unroll
            for (int r = 0; r < 8; ++r)
                partial[r] += __shfl_xor(partial[r], off, 32);

        if (lm == 0) {                              // lane 0 -> edges 0..7, lane 16 -> 8..15
            if (mlp == 0) {
                const float b2 = bw2[0];
#pragma unroll
                for (int r = 0; r < 8; ++r)
                    w_raw[tileBase + half * 8 + r] = partial[r] + b2;
            } else {
                const float b2 = ba2[0];
#pragma unroll
                for (int r = 0; r < 8; ++r) {
                    const int e = tileBase + half * 8 + r;
                    float a = __expf(partial[r] + b2);
                    a_ij[e] = a;
                    atomicAdd(&a_sum[(int)edges[e]], a);   // segment-sum over src
                }
            }
        }
    }
}

// ---------------------------------------------------------------------------
// Kernel 4: per-edge finalize + scatter of weighted unit differences.
// ---------------------------------------------------------------------------
__global__ void scatter_kernel(const float* __restrict__ x,
                               const long long* __restrict__ edges,
                               const float* __restrict__ w_raw,
                               const float* __restrict__ a_ij,
                               const float* __restrict__ a_sum,
                               float* __restrict__ out, int E) {
    int e = blockIdx.x * blockDim.x + threadIdx.x;
    if (e >= E) return;
    int s = (int)edges[e];
    int d = (int)edges[(size_t)E + e];
    float w = w_raw[e] * (a_ij[e] / a_sum[s]);
    float dx = x[d * 3 + 0] - x[s * 3 + 0];
    float dy = x[d * 3 + 1] - x[s * 3 + 1];
    float dz = x[d * 3 + 2] - x[s * 3 + 2];
    float nrm = sqrtf(dx * dx + dy * dy + dz * dz);
    nrm = fmaxf(nrm, 1e-12f);
    float sc = w / nrm;
    atomicAdd(&out[s * 3 + 0], dx * sc);
    atomicAdd(&out[s * 3 + 1], dy * sc);
    atomicAdd(&out[s * 3 + 2], dz * sc);
}

// ---------------------------------------------------------------------------
extern "C" void kernel_launch(void* const* d_in, const int* in_sizes, int n_in,
                              void* d_out, int out_size, void* d_ws, size_t ws_size,
                              hipStream_t stream) {
    const float*     x     = (const float*)d_in[0];
    const float*     m_ij  = (const float*)d_in[1];
    const long long* edges = (const long long*)d_in[2];   // int64 in reference
    const float*     Ww1   = (const float*)d_in[3];
    const float*     bw1   = (const float*)d_in[4];
    const float*     Ww2   = (const float*)d_in[5];
    const float*     bw2   = (const float*)d_in[6];
    const float*     Wa1   = (const float*)d_in[7];
    const float*     ba1   = (const float*)d_in[8];
    const float*     Wa2   = (const float*)d_in[9];
    const float*     ba2   = (const float*)d_in[10];
    float* out = (float*)d_out;

    const int N = in_sizes[0] / 3;     // 100000
    const int E = in_sizes[1] / 128;   // 1600000 (divisible by 16)

    // workspace layout
    char* ws = (char*)d_ws;
    __bf16* bufB  = (__bf16*)ws;                                       // 64 KB
    float*  w_raw = (float*)(ws + 65536);                              // E floats
    float*  aij   = (float*)(ws + 65536 + (size_t)E * 4);              // E floats
    float*  asum  = (float*)(ws + 65536 + (size_t)E * 8);              // N floats

    pack_weights_kernel<<<8, 256, 0, stream>>>(Ww1, Wa1, bufB);
    init_kernel<<<(3 * N + 255) / 256, 256, 0, stream>>>(x, out, asum, 3 * N, N);

    const int nTiles = E / 16;
    edge_mlp_kernel<<<(nTiles + 7) / 8, 256, 0, stream>>>(
        m_ij, bufB, bw1, Ww2, bw2, ba1, Wa2, ba2, edges, w_raw, aij, asum, nTiles);

    scatter_kernel<<<(E + 255) / 256, 256, 0, stream>>>(x, edges, w_raw, aij, asum, out, E);
}
